// DistanceFlowedAttention_15607911154067
// MI455X (gfx1250) — compile-verified, matrix operand order checked by reading, not run on hardware
//
#include <hip/hip_runtime.h>
#include <hip/hip_bf16.h>
#include <hip/hip_fp16.h>

// ---------------------------------------------------------------------------
// DistanceFlowedAttention for gfx1250 (MI455X), bf16 WMMA pipeline.
//   B=2, S=2048, D_MODEL=1024, H=16, Dh=64
// Stages:
//   1. coef  = sigmoid(inter_info @ We_w + b) + 1            (WMMA GEMM, f32 out)
//   2. Qp/Kp = coef * (X @ W)  -> bf16 [b,h,s,d]             (WMMA GEMM)
//   3. Vt    = (V @ W_V)^T     -> bf16 [b,h,d,s]             (WMMA GEMM)
//   4. fused attention: scores (WMMA) -> dist*mask (branchless) -> softmax in
//      256KB LDS -> attn written once to d_out -> context (WMMA) -> ctx bf16
//   5. out = ctx @ W_fc + input_Q                            (WMMA GEMM, f32 out)
// ---------------------------------------------------------------------------

typedef __attribute__((ext_vector_type(16))) __bf16 v16bf;
typedef __attribute__((ext_vector_type(8)))  float  v8f;

union FragBF {
    unsigned int   u32[8];
    uint4          q[2];
    v16bf          v;
};

// Native bf16 convert: let the backend use gfx1250 hardware cvt (+pk packing).
__device__ __forceinline__ unsigned short f2bf(float f) {
    return __builtin_bit_cast(unsigned short, (__bf16)f);
}
__device__ __forceinline__ unsigned int pack2(float lo, float hi) {
    return (unsigned int)f2bf(lo) | ((unsigned int)f2bf(hi) << 16);
}

#define MODE_COEF 0
#define MODE_QK   1
#define MODE_V    2
#define MODE_FC   3

// M=4096 (B*S), N=1024, K=1024 for every dense GEMM in this model.
// Block tile 128x128x32, 256 threads = 8 waves, wave tile 64x32 (4x2 WMMAs).
template <int MODE>
__global__ __launch_bounds__(256)
void gemm_kernel(const void* __restrict__ Aptr, const float* __restrict__ W,
                 const float* __restrict__ aux, void* __restrict__ outp)
{
    __shared__ unsigned short lA[128 * 32];   // [m][k] bf16
    __shared__ unsigned short lB[128 * 32];   // [n][k] bf16 (W transposed)

    const int tid   = threadIdx.x;
    const int lane  = tid & 31;
    const int wave  = tid >> 5;
    const int mbase = blockIdx.y * 128;
    const int nbase = blockIdx.x * 128;

    const int waveM = wave & 1;    // 2 groups of 64 rows
    const int waveN = wave >> 1;   // 4 groups of 32 cols

    v8f acc[4][2];
    #pragma unroll
    for (int i = 0; i < 4; i++)
        #pragma unroll
        for (int j = 0; j < 2; j++)
            acc[i][j] = v8f{0.f, 0.f, 0.f, 0.f, 0.f, 0.f, 0.f, 0.f};

    const int hrow  = lane & 15;
    const int khalf = (lane >> 4) * 8;

    for (int kb = 0; kb < 1024; kb += 32) {
        // prefetch next K-tile (emits global_prefetch_b8)
        if (kb + 32 < 1024) {
            if (MODE == MODE_FC) {
                const unsigned short* A = (const unsigned short*)Aptr;
                __builtin_prefetch(A + (size_t)(mbase + (tid >> 2)) * 1024 + kb + 32, 0, 0);
            } else {
                const float* A = (const float*)Aptr;
                __builtin_prefetch(A + (size_t)(mbase + (tid >> 3)) * 1024 + kb + 32, 0, 0);
            }
            __builtin_prefetch(W + (size_t)(kb + 32 + (tid >> 5)) * 1024 + nbase + (tid & 31) * 4, 0, 0);
        }
        // ---- stage A tile (128x32) into LDS as bf16 ----
        if (MODE == MODE_FC) {
            const unsigned short* A = (const unsigned short*)Aptr;  // already bf16
            const int c8 = (tid & 3) * 8;
            const int r0 = tid >> 2;                  // 0..63
            #pragma unroll
            for (int i = 0; i < 2; i++) {
                const int r = r0 + 64 * i;
                uint4 d = *(const uint4*)(A + (size_t)(mbase + r) * 1024 + kb + c8);
                *(uint4*)(&lA[r * 32 + c8]) = d;
            }
        } else {
            const float* A = (const float*)Aptr;
            const int c4 = (tid & 7) * 4;
            const int r0 = tid >> 3;                  // 0..31
            #pragma unroll
            for (int i = 0; i < 4; i++) {
                const int r = r0 + 32 * i;
                float4 d = *(const float4*)(A + (size_t)(mbase + r) * 1024 + kb + c4);
                unsigned int p0 = pack2(d.x, d.y);
                unsigned int p1 = pack2(d.z, d.w);
                *(uint2*)(&lA[r * 32 + c4]) = uint2{p0, p1};
            }
        }
        // ---- stage W tile (32x128) transposed into LDS [n][k] ----
        {
            const int c4 = (tid & 31) * 4;
            const int r0 = tid >> 5;                  // 0..7
            #pragma unroll
            for (int i = 0; i < 4; i++) {
                const int r = r0 + 8 * i;             // k row
                float4 d = *(const float4*)(W + (size_t)(kb + r) * 1024 + nbase + c4);
                lB[(c4 + 0) * 32 + r] = f2bf(d.x);
                lB[(c4 + 1) * 32 + r] = f2bf(d.y);
                lB[(c4 + 2) * 32 + r] = f2bf(d.z);
                lB[(c4 + 3) * 32 + r] = f2bf(d.w);
            }
        }
        __syncthreads();

        // ---- fragments: two b128 LDS reads per frag (ISA 16-bit A layout) ----
        FragBF a[4], b[2];
        #pragma unroll
        for (int i = 0; i < 4; i++) {
            const unsigned short* p = &lA[(waveM * 64 + i * 16 + hrow) * 32];
            a[i].q[0] = *(const uint4*)(p + khalf);
            a[i].q[1] = *(const uint4*)(p + 16 + khalf);
        }
        #pragma unroll
        for (int j = 0; j < 2; j++) {
            const unsigned short* p = &lB[(waveN * 32 + j * 16 + hrow) * 32];
            b[j].q[0] = *(const uint4*)(p + khalf);
            b[j].q[1] = *(const uint4*)(p + 16 + khalf);
        }
        #pragma unroll
        for (int i = 0; i < 4; i++)
            #pragma unroll
            for (int j = 0; j < 2; j++)
                acc[i][j] = __builtin_amdgcn_wmma_f32_16x16x32_bf16(
                    false, a[i].v, false, b[j].v, (short)0, acc[i][j], false, false);
        __syncthreads();
    }

    // ---- epilogue ----
    const int rofs = (lane >> 4) * 8;
    #pragma unroll
    for (int i = 0; i < 4; i++) {
        #pragma unroll
        for (int j = 0; j < 2; j++) {
            #pragma unroll
            for (int r = 0; r < 8; r++) {
                const int m = mbase + waveM * 64 + i * 16 + rofs + r;
                const int n = nbase + waveN * 32 + j * 16 + hrow;
                float v = acc[i][j][r];
                if (MODE == MODE_COEF) {
                    v += aux[n];                                     // bias
                    v = 1.f / (1.f + __expf(-v)) + 1.f;              // sigmoid + 1
                    ((float*)outp)[(size_t)m * 1024 + n] = v;
                } else if (MODE == MODE_QK) {
                    v *= aux[(size_t)m * 1024 + n];                  // coef gate
                    const int bb = m >> 11, s = m & 2047, h = n >> 6, d = n & 63;
                    ((unsigned short*)outp)[(((size_t)(bb * 16 + h) * 2048) + s) * 64 + d] = f2bf(v);
                } else if (MODE == MODE_V) {
                    const int bb = m >> 11, s = m & 2047, h = n >> 6, d = n & 63;
                    ((unsigned short*)outp)[(((size_t)(bb * 16 + h) * 64) + d) * 2048 + s] = f2bf(v);
                } else { // MODE_FC
                    v += aux[(size_t)m * 1024 + n];                  // residual input_Q
                    ((float*)outp)[(size_t)m * 1024 + n] = v;
                }
            }
        }
    }
}

// ---------------------------------------------------------------------------
// Fused attention: one workgroup per (b, h, 32-row stripe).
// LDS: 32x2048 f32 scores (256 KB) + 32x64 bf16 Q tile + reduce scratch.
// ---------------------------------------------------------------------------
__global__ __launch_bounds__(256)
void attn_kernel(const unsigned short* __restrict__ Qp,
                 const unsigned short* __restrict__ Kp,
                 const unsigned short* __restrict__ Vt,
                 const float* __restrict__ dist,
                 const unsigned char* __restrict__ mask,
                 float* __restrict__ attn_out,
                 unsigned short* __restrict__ ctx)
{
    extern __shared__ unsigned char smem[];
    float*          SC   = (float*)smem;                                // [32][2048]
    unsigned short* QT   = (unsigned short*)(smem + 32 * 2048 * 4);     // [32][64]
    float*          RED  = (float*)(smem + 32 * 2048 * 4 + 32 * 64 * 2);// [256]
    float*          ROWV = RED + 256;                                   // [32]

    const int tid  = threadIdx.x, lane = tid & 31, wave = tid >> 5;
    const int blk  = blockIdx.x;          // 0..2047
    const int sblk = blk & 63;
    const int h    = (blk >> 6) & 15;
    const int b    = blk >> 10;
    const int bh   = b * 16 + h;
    const int s0   = sblk * 32;

    const int hrow  = lane & 15;
    const int khalf = (lane >> 4) * 8;
    const int rofs  = (lane >> 4) * 8;

    // Q tile -> LDS (already bf16, 2048 ushorts)
    ((uint4*)QT)[tid] = ((const uint4*)(Qp + ((size_t)bh * 2048 + s0) * 64))[tid];
    __syncthreads();

    // ---- scores: each wave owns 256 columns (16 col-tiles) ----
    FragBF af[2][2];
    #pragma unroll
    for (int rt = 0; rt < 2; rt++)
        #pragma unroll
        for (int kk = 0; kk < 2; kk++) {
            const unsigned short* p = &QT[(rt * 16 + hrow) * 64 + kk * 32];
            af[rt][kk].q[0] = *(const uint4*)(p + khalf);
            af[rt][kk].q[1] = *(const uint4*)(p + 16 + khalf);
        }
    const int colbase = wave * 256;
    for (int ct = 0; ct < 16; ct++) {
        const int ttile = colbase + ct * 16;
        FragBF bf[2];
        #pragma unroll
        for (int kk = 0; kk < 2; kk++) {
            const unsigned short* p = Kp + ((size_t)bh * 2048 + ttile + hrow) * 64 + kk * 32;
            bf[kk].q[0] = *(const uint4*)(p + khalf);
            bf[kk].q[1] = *(const uint4*)(p + 16 + khalf);
        }
        #pragma unroll
        for (int rt = 0; rt < 2; rt++) {
            v8f acc = v8f{0.f, 0.f, 0.f, 0.f, 0.f, 0.f, 0.f, 0.f};
            acc = __builtin_amdgcn_wmma_f32_16x16x32_bf16(false, af[rt][0].v, false, bf[0].v, (short)0, acc, false, false);
            acc = __builtin_amdgcn_wmma_f32_16x16x32_bf16(false, af[rt][1].v, false, bf[1].v, (short)0, acc, false, false);
            const int t = ttile + hrow;
            // Branchless epilogue: load dist+mask unconditionally, then select.
            const size_t dbase = ((size_t)b * 2048 + (s0 + rt * 16 + rofs)) * 2048 + t;
            float         dv[8];
            unsigned char mv[8];
            #pragma unroll
            for (int r = 0; r < 8; r++) {
                dv[r] = dist[dbase + (size_t)r * 2048];
                mv[r] = mask[dbase + (size_t)r * 2048];
            }
            #pragma unroll
            for (int r = 0; r < 8; r++) {
                const int srow = rt * 16 + rofs + r;
                float v = dv[r] * (acc[r] * 0.125f);
                v = mv[r] ? -1e10f : v;                 // v_cndmask, no branch
                SC[srow * 2048 + t] = v;
            }
        }
    }
    __syncthreads();

    // ---- softmax over each of 32 rows, 8 threads per row ----
    const int row = tid >> 3;
    const int sub = tid & 7;
    float pmax = -3.4e38f;
    for (int c = sub; c < 2048; c += 8) pmax = fmaxf(pmax, SC[row * 2048 + c]);
    RED[tid] = pmax;
    __syncthreads();
    float rmax = RED[row * 8];
    #pragma unroll
    for (int i = 1; i < 8; i++) rmax = fmaxf(rmax, RED[row * 8 + i]);
    __syncthreads();
    float psum = 0.f;
    for (int c = sub; c < 2048; c += 8) {
        const float e = __expf(SC[row * 2048 + c] - rmax);
        SC[row * 2048 + c] = e;
        psum += e;
    }
    RED[tid] = psum;
    __syncthreads();
    if (sub == 0) {
        float s = 0.f;
        #pragma unroll
        for (int i = 0; i < 8; i++) s += RED[row * 8 + i];
        ROWV[row] = 1.f / s;
    }
    __syncthreads();

    // ---- normalize + single coalesced attn write to HBM ----
    float* aout = attn_out + (size_t)bh * 2048 * 2048 + (size_t)s0 * 2048;
    for (int idx = tid; idx < 32 * 2048; idx += 256) {
        const int rr = idx >> 11;
        const float v = SC[idx] * ROWV[rr];
        SC[idx]   = v;
        aout[idx] = v;
    }
    __syncthreads();

    // ---- context: 32x64 = 8 tiles, one per wave; K loop over t (2048) ----
    {
        const int rt = wave & 1, ctile = wave >> 1;
        v8f acc = v8f{0.f, 0.f, 0.f, 0.f, 0.f, 0.f, 0.f, 0.f};
        const unsigned short* vrow = Vt + ((size_t)bh * 64 + ctile * 16 + hrow) * 2048;
        for (int kt = 0; kt < 2048; kt += 32) {
            FragBF a;
            const float* ap = &SC[(rt * 16 + hrow) * 2048 + kt];
            const float4 lo0 = *(const float4*)(ap + khalf);
            const float4 lo1 = *(const float4*)(ap + khalf + 4);
            const float4 hi0 = *(const float4*)(ap + 16 + khalf);
            const float4 hi1 = *(const float4*)(ap + 16 + khalf + 4);
            a.u32[0] = pack2(lo0.x, lo0.y); a.u32[1] = pack2(lo0.z, lo0.w);
            a.u32[2] = pack2(lo1.x, lo1.y); a.u32[3] = pack2(lo1.z, lo1.w);
            a.u32[4] = pack2(hi0.x, hi0.y); a.u32[5] = pack2(hi0.z, hi0.w);
            a.u32[6] = pack2(hi1.x, hi1.y); a.u32[7] = pack2(hi1.z, hi1.w);
            FragBF bb;
            bb.q[0] = *(const uint4*)(vrow + kt + khalf);
            bb.q[1] = *(const uint4*)(vrow + kt + 16 + khalf);
            acc = __builtin_amdgcn_wmma_f32_16x16x32_bf16(false, a.v, false, bb.v, (short)0, acc, false, false);
        }
        #pragma unroll
        for (int r = 0; r < 8; r++) {
            const int sg   = s0 + rt * 16 + rofs + r;
            const int dcol = h * 64 + ctile * 16 + hrow;
            ctx[((size_t)b * 2048 + sg) * 1024 + dcol] = f2bf(acc[r]);
        }
    }
}

// ---------------------------------------------------------------------------
extern "C" void kernel_launch(void* const* d_in, const int* in_sizes, int n_in,
                              void* d_out, int out_size, void* d_ws, size_t ws_size,
                              hipStream_t stream) {
    (void)in_sizes; (void)n_in; (void)out_size; (void)ws_size;
    const float* input_Q    = (const float*)d_in[0];
    const float* input_K    = (const float*)d_in[1];
    const float* input_V    = (const float*)d_in[2];
    const float* inter_info = (const float*)d_in[3];
    const float* dist       = (const float*)d_in[4];
    const unsigned char* attn_mask = (const unsigned char*)d_in[5];  // jnp.bool_ = 1 byte
    const float* W_Q  = (const float*)d_in[6];
    const float* W_K  = (const float*)d_in[7];
    const float* W_V  = (const float*)d_in[8];
    const float* W_fc = (const float*)d_in[9];
    const float* We_w = (const float*)d_in[10];
    const float* We_b = (const float*)d_in[11];

    // Workspace layout (48 MB total)
    char* ws = (char*)d_ws;
    float*          coef = (float*)ws;                                       // 16 MB
    unsigned short* Qp   = (unsigned short*)(ws + (size_t)16 * 1024 * 1024); // 8 MB
    unsigned short* Kp   = (unsigned short*)(ws + (size_t)24 * 1024 * 1024); // 8 MB
    unsigned short* Vt   = (unsigned short*)(ws + (size_t)32 * 1024 * 1024); // 8 MB
    unsigned short* ctx  = (unsigned short*)(ws + (size_t)40 * 1024 * 1024); // 8 MB

    float* out_main = (float*)d_out;                       // [2,2048,1024]
    float* attn_out = out_main + (size_t)2 * 2048 * 1024;  // [2,16,2048,2048]

    const dim3 ggrid(8, 32);   // N/128 x M/128
    const dim3 gblk(256);

    // 1. coef = sigmoid(inter_info @ We_w + b) + 1
    hipLaunchKernelGGL((gemm_kernel<MODE_COEF>), ggrid, gblk, 0, stream,
                       (const void*)inter_info, We_w, We_b, (void*)coef);
    // 2. Qp / Kp projections with coef gating
    hipLaunchKernelGGL((gemm_kernel<MODE_QK>), ggrid, gblk, 0, stream,
                       (const void*)input_Q, W_Q, coef, (void*)Qp);
    hipLaunchKernelGGL((gemm_kernel<MODE_QK>), ggrid, gblk, 0, stream,
                       (const void*)input_K, W_K, coef, (void*)Kp);
    // 3. V projection, stored transposed per head
    hipLaunchKernelGGL((gemm_kernel<MODE_V>), ggrid, gblk, 0, stream,
                       (const void*)input_V, W_V, (const float*)nullptr, (void*)Vt);
    // 4. fused attention (needs 261 KB dynamic LDS; CDNA5 WGP has 320 KB)
    const size_t smem = (size_t)32 * 2048 * 4 + 32 * 64 * 2 + 256 * 4 + 32 * 4;
    (void)hipFuncSetAttribute((const void*)attn_kernel,
                              hipFuncAttributeMaxDynamicSharedMemorySize, (int)smem);
    hipLaunchKernelGGL(attn_kernel, dim3(2048), gblk, smem, stream,
                       Qp, Kp, Vt, dist, attn_mask, attn_out, ctx);
    // 5. output = ctx @ W_fc + input_Q
    hipLaunchKernelGGL((gemm_kernel<MODE_FC>), ggrid, gblk, 0, stream,
                       (const void*)ctx, W_fc, input_Q, (void*)out_main);
}
